// MoEMLPBlock_34153579938013
// MI455X (gfx1250) — matrix operand order repaired and checked by reading, not compiled, not run on hardware
//
#include <hip/hip_runtime.h>
#include <math.h>

// ---------------------------------------------------------------------------
// Soft-MoE linear for gfx1250 (CDNA5, wave32):
//   kernel 1/2: one-time f32->bf16 conversion of w and x into workspace
//   kernel 3  : GEMM  y[b,j] = sum_e coef[b,e] * (x[b,:] . w[e,j,:])
//               - 128x128 block tile, 8 waves (4M x 2N), 2x4 WMMA tiles/wave
//               - 128x512 bf16 x-slab resident in LDS (async global->LDS copy)
//               - per-expert f32 partials, coef folded at combine time
//   kernel 4  : in-place bias + LayerNorm + exact-erf GELU per row
// ---------------------------------------------------------------------------

typedef __bf16 bf16;
typedef bf16  v16bf __attribute__((ext_vector_type(16)));
typedef bf16  v8bf  __attribute__((ext_vector_type(8)));
typedef bf16  v4bf  __attribute__((ext_vector_type(4)));
typedef float v8f   __attribute__((ext_vector_type(8)));
typedef float v4f   __attribute__((ext_vector_type(4)));
typedef int   gv4i  __attribute__((vector_size(4 * sizeof(int))));  // builtin's b128 unit

#define B_TOK   16384
#define IN_DIM  1024
#define OUT_DIM 1024
#define NEXP    8
#define KSEG    512               // k-columns resident in LDS per stage
#define LDS_STRIDE (KSEG + 8)     // pad: 260 dwords/row -> 4-bank rotation

#define GAS __attribute__((address_space(1)))
#define LAS __attribute__((address_space(3)))

#if defined(__AMDGCN__) && __has_builtin(__builtin_amdgcn_global_load_async_to_lds_b128)
#define HAVE_ASYNC 1
#else
#define HAVE_ASYNC 0
#endif

static __device__ __forceinline__ void wait_async0() {
#if defined(__AMDGCN__)
#if __has_builtin(__builtin_amdgcn_s_wait_asynccnt)
    __builtin_amdgcn_s_wait_asynccnt(0);
#else
    asm volatile("s_wait_asynccnt 0x0" ::: "memory");
#endif
#endif
}

// f32 -> bf16 round-to-nearest-even
static __device__ __forceinline__ bf16 f2bf(float f) {
    unsigned u = __builtin_bit_cast(unsigned, f);
    unsigned r = u + 0x7FFFu + ((u >> 16) & 1u);
    unsigned short h = (unsigned short)(r >> 16);
    return __builtin_bit_cast(bf16, h);
}

// ---------------------------------------------------------------------------
// Kernel 1: elementwise f32 -> bf16 (used for both w and x)
// ---------------------------------------------------------------------------
__global__ __launch_bounds__(256) void cvt_bf16(const float* __restrict__ in,
                                                bf16* __restrict__ out, int n) {
    int i = (blockIdx.x * blockDim.x + threadIdx.x) * 4;
    if (i < n) {
        v4f f = *(const v4f*)(in + i);
        v4bf o4;
        o4[0] = f2bf(f[0]); o4[1] = f2bf(f[1]);
        o4[2] = f2bf(f[2]); o4[3] = f2bf(f[3]);
        *(v4bf*)(out + i) = o4;
    }
}

// ---------------------------------------------------------------------------
// Kernel 2: WMMA GEMM with LDS-resident x slab
// ---------------------------------------------------------------------------
__global__ __launch_bounds__(256) void moe_gemm(const bf16* __restrict__ xbf,
                                                const float* __restrict__ coef,
                                                const bf16* __restrict__ wbf,
                                                float* __restrict__ y) {
    __shared__ bf16  ldsX[128][LDS_STRIDE];   // ~133 KB
    __shared__ float ldsC[128][NEXP];         // 4 KB coef tile

    const int rowBase = blockIdx.x * 128;     // M
    const int colBase = blockIdx.y * 128;     // N
    const int tid  = threadIdx.x;
    const int lane = tid & 31;
    const int wave = tid >> 5;
    const int waveM = wave & 3;               // 0..3 : 32 rows each
    const int waveN = wave >> 2;              // 0..1 : 64 cols each
    const int m16  = lane & 15;
    const int half = lane >> 4;

    // ---- coef tile -> LDS (once per block) ----
    {
        const int r = tid >> 1, p = tid & 1;
        v4f c4 = *(const v4f*)(coef + (size_t)(rowBase + r) * NEXP + p * 4);
        *(v4f*)&ldsC[r][p * 4] = c4;
    }

    v8f acc[2][4];
#pragma unroll
    for (int mt = 0; mt < 2; ++mt)
#pragma unroll
        for (int nt = 0; nt < 4; ++nt)
#pragma unroll
            for (int i = 0; i < 8; ++i) acc[mt][nt][i] = 0.0f;

    for (int kseg = 0; kseg < IN_DIM / KSEG; ++kseg) {
        __syncthreads();  // previous slab's readers done

        // ---- stage x slab (128 x KSEG bf16) global -> LDS ----
        {
            const int r = tid >> 1;
            const int s = (tid & 1) * (KSEG / 2);
            const bf16* gsrc = xbf + (size_t)(rowBase + r) * IN_DIM +
                               kseg * KSEG + s;
            bf16* ldst = &ldsX[r][s];
#if HAVE_ASYNC
#pragma unroll
            for (int c = 0; c < KSEG / 2; c += 8)
                __builtin_amdgcn_global_load_async_to_lds_b128(
                    (GAS gv4i*)(gsrc + c), (LAS gv4i*)(ldst + c), 0, 0);
#else
#pragma unroll 8
            for (int c = 0; c < KSEG / 2; c += 8)
                *(v8bf*)(ldst + c) = *(const v8bf*)(gsrc + c);
#endif
        }
#if HAVE_ASYNC
        wait_async0();
#endif
        __syncthreads();

        // ---- per expert: accumulate partial over this k slab, then combine ----
        for (int e = 0; e < NEXP; ++e) {
            v8f par[2][4];
#pragma unroll
            for (int mt = 0; mt < 2; ++mt)
#pragma unroll
                for (int nt = 0; nt < 4; ++nt)
#pragma unroll
                    for (int i = 0; i < 8; ++i) par[mt][nt][i] = 0.0f;

            const bf16* wseg = wbf + (size_t)e * OUT_DIM * IN_DIM +
                               kseg * KSEG;

#pragma unroll 2
            for (int k0 = 0; k0 < KSEG; k0 += 32) {
                // A fragments from LDS (contiguous k order, matching B):
                // lane<16: K 0..15 ; lane>=16: K 16..31
                v16bf aF[2];
#pragma unroll
                for (int mt = 0; mt < 2; ++mt) {
                    const bf16* ap = &ldsX[waveM * 32 + mt * 16 + m16][k0];
                    v8bf lo = *(const v8bf*)(ap + half * 16);
                    v8bf hi = *(const v8bf*)(ap + half * 16 + 8);
                    aF[mt] = __builtin_shufflevector(lo, hi,
                              0, 1, 2, 3, 4, 5, 6, 7,
                              8, 9, 10, 11, 12, 13, 14, 15);
                }
                // B fragments direct from global (k-contiguous rows of w)
                const bf16* wk = wseg + k0 + half * 16;
#pragma unroll
                for (int nt = 0; nt < 4; ++nt) {
                    const int col = colBase + waveN * 64 + nt * 16 + m16;
                    v16bf bF = *(const v16bf*)(wk + (size_t)col * IN_DIM);
#pragma unroll
                    for (int mt = 0; mt < 2; ++mt)
                        par[mt][nt] = __builtin_amdgcn_wmma_f32_16x16x32_bf16(
                            false, aF[mt], false, bF, (short)0, par[mt][nt],
                            false, false);
                }
            }

            // combine: acc += coef[row, e] * partial
#pragma unroll
            for (int mt = 0; mt < 2; ++mt) {
                const int rb = waveM * 32 + mt * 16 + half * 8;
                float ce[8];
#pragma unroll
                for (int v = 0; v < 8; ++v) ce[v] = ldsC[rb + v][e];
#pragma unroll
                for (int nt = 0; nt < 4; ++nt)
#pragma unroll
                    for (int v = 0; v < 8; ++v)
                        acc[mt][nt][v] += ce[v] * par[mt][nt][v];
            }
        }
    }

    // ---- epilogue: C layout (VGPR v -> M = v + 8*half, N = m16) ----
#pragma unroll
    for (int mt = 0; mt < 2; ++mt) {
#pragma unroll
        for (int nt = 0; nt < 4; ++nt) {
            const int r0 = rowBase + waveM * 32 + mt * 16 + half * 8;
            const int c  = colBase + waveN * 64 + nt * 16 + m16;
#pragma unroll
            for (int v = 0; v < 8; ++v)
                y[(size_t)(r0 + v) * OUT_DIM + c] = acc[mt][nt][v];
        }
    }
}

// ---------------------------------------------------------------------------
// Kernel 3: in-place  y = GELU(LayerNorm(y + coef@b))  per row
// ---------------------------------------------------------------------------
__global__ __launch_bounds__(256) void ln_gelu(float* __restrict__ y,
                                               const float* __restrict__ coef,
                                               const float* __restrict__ bvec,
                                               const float* __restrict__ gamma,
                                               const float* __restrict__ beta) {
    __shared__ float sred[256];
    __shared__ float sred2[256];
    const int row = blockIdx.x;
    const int tid = threadIdx.x;

    float c[NEXP];
#pragma unroll
    for (int e = 0; e < NEXP; ++e) c[e] = coef[(size_t)row * NEXP + e];

    float v[4];
    float s = 0.f, q = 0.f;
#pragma unroll
    for (int i = 0; i < 4; ++i) {
        const int j = i * 256 + tid;
        float t = y[(size_t)row * OUT_DIM + j];
        float bias = 0.f;
#pragma unroll
        for (int e = 0; e < NEXP; ++e) bias += c[e] * bvec[e * OUT_DIM + j];
        t += bias;
        v[i] = t; s += t; q += t * t;
    }
    sred[tid] = s; sred2[tid] = q;
    __syncthreads();
#pragma unroll
    for (int off = 128; off > 0; off >>= 1) {
        if (tid < off) {
            sred[tid]  += sred[tid + off];
            sred2[tid] += sred2[tid + off];
        }
        __syncthreads();
    }
    const float mean = sred[0]  * (1.0f / OUT_DIM);
    const float var  = sred2[0] * (1.0f / OUT_DIM) - mean * mean;
    const float rstd = rsqrtf(var + 1e-5f);
#pragma unroll
    for (int i = 0; i < 4; ++i) {
        const int j = i * 256 + tid;
        float t = (v[i] - mean) * rstd * gamma[j] + beta[j];
        float g = 0.5f * t * (1.0f + erff(t * 0.70710678118654752f));
        y[(size_t)row * OUT_DIM + j] = g;
    }
}

// ---------------------------------------------------------------------------
extern "C" void kernel_launch(void* const* d_in, const int* in_sizes, int n_in,
                              void* d_out, int out_size, void* d_ws, size_t ws_size,
                              hipStream_t stream) {
    const float* x     = (const float*)d_in[0];  // [B, IN]
    const float* coef  = (const float*)d_in[1];  // [B, E]
    const float* w     = (const float*)d_in[2];  // [E, OUT, IN]
    const float* bvec  = (const float*)d_in[3];  // [E, OUT]
    const float* gamma = (const float*)d_in[4];  // [OUT]
    const float* beta  = (const float*)d_in[5];  // [OUT]
    float* y  = (float*)d_out;                   // [B, OUT]

    bf16* wbf = (bf16*)d_ws;                               // 16 MB
    bf16* xbf = wbf + (size_t)NEXP * OUT_DIM * IN_DIM;     // +32 MB

    (void)in_sizes; (void)n_in; (void)out_size; (void)ws_size;

    const int wn = NEXP * OUT_DIM * IN_DIM;
    const int xn = B_TOK * IN_DIM;
    cvt_bf16<<<wn / (256 * 4), 256, 0, stream>>>(w, wbf, wn);
    cvt_bf16<<<xn / (256 * 4), 256, 0, stream>>>(x, xbf, xn);

    dim3 grid(B_TOK / 128, OUT_DIM / 128);
    moe_gemm<<<grid, 256, 0, stream>>>(xbf, coef, wbf, y);

    ln_gelu<<<B_TOK, 256, 0, stream>>>(y, coef, bvec, gamma, beta);
}
